// GraphAttention_7224134992064
// MI455X (gfx1250) — compile-verified, matrix operand order checked by reading, not run on hardware
//
#include <hip/hip_runtime.h>
#include <hip/hip_bf16.h>

// ---------------------------------------------------------------------------
// Problem dims (hardcoded from reference)
// ---------------------------------------------------------------------------
#define BB   4
#define NN   2048
#define DD   512
#define OUTD 512
#define HH   8
#define HDD  64
#define MTOT (BB * NN)          // 8192 rows
#define SCALE 0.125f            // HD^-0.5 = 1/8

typedef __attribute__((ext_vector_type(16))) __bf16 bf16x16;
typedef __attribute__((ext_vector_type(8)))  float  f32x8;
typedef __attribute__((ext_vector_type(4)))  unsigned int u32x4;
typedef __attribute__((ext_vector_type(4)))  int          i32x4;

// ---------------------------------------------------------------------------
// gfx1250 async global->LDS copy (ASYNCcnt path, CDNA5 ISA).
// Builtin expects (AS1 i32x4* gsrc, AS3 i32x4* ldst, imm offset, imm cpol).
// Guarded: falls back to a synchronous b128 load + ds_store if unavailable.
// ---------------------------------------------------------------------------
#if __has_builtin(__builtin_amdgcn_global_load_async_to_lds_b128)
#define HAVE_ASYNC_LDS 1
#else
#define HAVE_ASYNC_LDS 0
#endif

__device__ __forceinline__ void cp16_g2l(const __bf16* g, __bf16* l) {
#if HAVE_ASYNC_LDS
    __builtin_amdgcn_global_load_async_to_lds_b128(
        (__attribute__((address_space(1))) i32x4*)(unsigned long long)g,
        (__attribute__((address_space(3))) i32x4*)(unsigned long long)l,
        0, 0);
#else
    *reinterpret_cast<u32x4*>(l) = *reinterpret_cast<const u32x4*>(g);
#endif
}

__device__ __forceinline__ void async_wait_all() {
#if HAVE_ASYNC_LDS
    asm volatile("s_wait_asynccnt 0" ::: "memory");
#endif
}

// ---------------------------------------------------------------------------
// Load one WMMA A/B operand (16 bf16) for this lane.
// Per ISA layout, lane-half 0 holds K = {0..7, 16..23}, half 1 holds
// K = {8..15, 24..31} of its row/column: two 16-byte contiguous chunks.
// Works for both global and LDS pointers.
// ---------------------------------------------------------------------------
__device__ __forceinline__ bf16x16 load_ab(const __bf16* base, int half) {
    union { u32x4 u[2]; bf16x16 v; } r;
    r.u[0] = *reinterpret_cast<const u32x4*>(base + half * 8);
    r.u[1] = *reinterpret_cast<const u32x4*>(base + 16 + half * 8);
    return r.v;
}

__device__ __forceinline__ f32x8 wmma_bf16(bf16x16 a, bf16x16 b, f32x8 c) {
    return __builtin_amdgcn_wmma_f32_16x16x32_bf16(
        false, a, false, b, (short)0, c, false, false);
}

// ---------------------------------------------------------------------------
// Conversions
// ---------------------------------------------------------------------------
__global__ void cvt_f32_bf16(const float* __restrict__ src,
                             __bf16* __restrict__ dst, int n) {
    int i = blockIdx.x * blockDim.x + threadIdx.x;
    if (i < n) dst[i] = (__bf16)src[i];
}

// W[R,C] fp32 -> Wt[C,R] bf16 (transpose so B-matrix columns are K-contig)
__global__ void cvt_transpose_bf16(const float* __restrict__ W,
                                   __bf16* __restrict__ Wt, int R, int C) {
    int i = blockIdx.x * blockDim.x + threadIdx.x;
    if (i < R * C) {
        int r = i / C, c = i % C;
        Wt[(size_t)c * R + r] = (__bf16)W[i];
    }
}

// ---------------------------------------------------------------------------
// GEMM: Out[M,512] = A[M,512] @ Bt^T + bias.  A row-major bf16, Bt is the
// 512x512 weight stored transposed (col-major of original) bf16.
// 256 threads = 8 waves; each wave owns a 16x16 output tile; block = 64x32.
// storeMode: 0 = bf16 head-split [B,H,N,HD]
//            1 = bf16 head-split transposed [B,H,HD,N]   (for V)
//            2 = fp32 row-major [M,512]                  (final output)
// ---------------------------------------------------------------------------
__global__ void gemm512_bf16(const __bf16* __restrict__ A,
                             const __bf16* __restrict__ Bt,
                             const float*  __restrict__ bias,
                             void* __restrict__ Out, int storeMode) {
    const int lane = threadIdx.x & 31;
    const int wv   = threadIdx.x >> 5;
    const int half = lane >> 4;
    const int l16  = lane & 15;

    const int mbase = blockIdx.x * 64 + (wv & 3) * 16;
    const int nbase = blockIdx.y * 32 + (wv >> 2) * 16;

    const __bf16* Ar = A  + (size_t)(mbase + l16) * DD;
    const __bf16* Bc = Bt + (size_t)(nbase + l16) * DD;

    f32x8 acc = {};
#pragma unroll 4
    for (int k = 0; k < DD; k += 32) {
        __builtin_prefetch(Ar + k + 128, 0, 1);   // global_prefetch_b8
        bf16x16 a = load_ab(Ar + k, half);
        bf16x16 b = load_ab(Bc + k, half);
        acc = wmma_bf16(a, b, acc);
    }

    const int n   = nbase + l16;
    const float bv = bias[n];
    const int hh = n >> 6, hd = n & 63;

#pragma unroll
    for (int j = 0; j < 8; ++j) {
        int m  = mbase + half * 8 + j;
        float v = acc[j] + bv;
        int bb = m >> 11, nr = m & (NN - 1);
        if (storeMode == 2) {
            ((float*)Out)[(size_t)m * OUTD + n] = v;
        } else if (storeMode == 0) {
            ((__bf16*)Out)[(((size_t)bb * HH + hh) * NN + nr) * HDD + hd] = (__bf16)v;
        } else {
            ((__bf16*)Out)[(((size_t)bb * HH + hh) * HDD + hd) * NN + nr] = (__bf16)v;
        }
    }
}

// ---------------------------------------------------------------------------
// Fused flash-attention with adjacency mask.
// Block = 128 threads (4 waves); all 4 waves share one (b,h) and adjacent
// 16-query tiles, so the K/V sweep is staged ONCE per block into LDS via
// double-buffered async global->LDS copies (ASYNCcnt), then consumed by all
// waves through ds_load_b128.  Per 32-key step: S = Q K^T (4 WMMA), mask +
// online softmax (shfl_xor row reductions within 16-lane halves), P relayout
// C->A through LDS, O += P V (4 WMMA, V pre-transposed [B,H,HD,N]).
// ---------------------------------------------------------------------------
__global__ void flash_attn_bf16(const __bf16* __restrict__ Q,
                                const __bf16* __restrict__ K,
                                const __bf16* __restrict__ Vt,
                                const int*    __restrict__ adj,
                                __bf16* __restrict__ Oout) {
    __shared__ __bf16 kbuf[2][32 * HDD];   // 2 x 4 KB : 32 keys x 64 hd
    __shared__ __bf16 vbuf[2][HDD * 32];   // 2 x 4 KB : 64 hd  x 32 keys
    __shared__ __bf16 pbuf[4][16 * 32];    // per-wave P relayout tile

    const int tid  = threadIdx.x;
    const int lane = tid & 31;
    const int wv   = tid >> 5;
    const int half = lane >> 4;
    const int l16  = lane & 15;

    const int b = blockIdx.z, h = blockIdx.y;
    const int qbase = (blockIdx.x * 4 + wv) * 16;

    const __bf16* Qb = Q  + (((size_t)b * HH + h) * NN + qbase) * HDD;
    const __bf16* Kb = K  + ((size_t)b * HH + h) * NN * HDD;
    const __bf16* Vb = Vt + ((size_t)b * HH + h) * HDD * NN;
    const int* adjr  = adj + (size_t)(qbase + half * 8) * NN;

    // Stage one 32-key K tile + V tile into LDS buffer `buf`.
    // K tile is 4 KB contiguous in memory (keys x 64 bf16 rows).
    // V tile is 64 rows (hd) of 64 B, row stride NN*2 bytes.
    auto stage = [&](int buf, int kb0) {
#pragma unroll
        for (int o = 0; o < 2; ++o) {
            int u = tid * 2 + o;                       // 256 x 16B units
            cp16_g2l(Kb + (size_t)kb0 * HDD + u * 8, &kbuf[buf][u * 8]);
        }
#pragma unroll
        for (int o = 0; o < 2; ++o) {
            int u = tid * 2 + o;                       // 256 x 16B units
            int row = u >> 2, ch = u & 3;              // 64 rows x 4 chunks
            cp16_g2l(Vb + (size_t)row * NN + kb0 + ch * 8,
                     &vbuf[buf][row * 32 + ch * 8]);
        }
    };

    // Q tile 16x64 as two A operands (K = 0..31, 32..63), held in registers
    const bf16x16 qa0 = load_ab(Qb + l16 * HDD + 0,  half);
    const bf16x16 qa1 = load_ab(Qb + l16 * HDD + 32, half);

    f32x8 acc[4] = {};
    float mrow[8], lrow[8];
#pragma unroll
    for (int j = 0; j < 8; ++j) { mrow[j] = -1e30f; lrow[j] = 0.0f; }

    __bf16* lp = &pbuf[wv][0];

    stage(0, 0);   // prologue: fill buffer 0

    for (int kb0 = 0; kb0 < NN; kb0 += 32) {
        const int cur = (kb0 >> 5) & 1;

        // My async copies for buf[cur] done; all waves past reads of the
        // buffer we are about to overwrite.
        async_wait_all();
        __syncthreads();

        if (kb0 + 32 < NN) stage(cur ^ 1, kb0 + 32);   // overlap next tile

        const __bf16* ldsK = &kbuf[cur][0];
        const __bf16* ldsV = &vbuf[cur][0];

        // ---- S = Q @ K^T for 32 keys (two 16x16 C tiles), B from LDS ----
        f32x8 s0 = {}, s1 = {};
        s0 = wmma_bf16(qa0, load_ab(ldsK + l16 * HDD,             half), s0);
        s0 = wmma_bf16(qa1, load_ab(ldsK + l16 * HDD + 32,        half), s0);
        s1 = wmma_bf16(qa0, load_ab(ldsK + (16 + l16) * HDD,      half), s1);
        s1 = wmma_bf16(qa1, load_ab(ldsK + (16 + l16) * HDD + 32, half), s1);

        // ---- mask + online softmax (rows live per lane-half) ----
        float p0[8], p1[8];
        const int c0 = kb0 + l16, c1 = kb0 + 16 + l16;
#pragma unroll
        for (int j = 0; j < 8; ++j) {
            float v0 = s0[j] * SCALE;
            float v1 = s1[j] * SCALE;
            if (adjr[(size_t)j * NN + c0] == 0) v0 = -1e9f;
            if (adjr[(size_t)j * NN + c1] == 0) v1 = -1e9f;

            float rm = fmaxf(v0, v1);
#pragma unroll
            for (int off = 1; off < 16; off <<= 1)
                rm = fmaxf(rm, __shfl_xor(rm, off, 16));

            float mn = fmaxf(mrow[j], rm);
            float al = __expf(mrow[j] - mn);
            float e0 = __expf(v0 - mn);
            float e1 = __expf(v1 - mn);
            float rs = e0 + e1;
#pragma unroll
            for (int off = 1; off < 16; off <<= 1)
                rs += __shfl_xor(rs, off, 16);

            lrow[j] = lrow[j] * al + rs;
            mrow[j] = mn;
            p0[j] = e0; p1[j] = e1;
            acc[0][j] *= al; acc[1][j] *= al;
            acc[2][j] *= al; acc[3][j] *= al;
        }

        // ---- P: C-layout -> LDS -> A-layout (per-wave region) ----
#pragma unroll
        for (int j = 0; j < 8; ++j) {
            int r = half * 8 + j;
            lp[r * 32 + l16]      = (__bf16)p0[j];
            lp[r * 32 + 16 + l16] = (__bf16)p1[j];
        }
        asm volatile("s_wait_dscnt 0" ::: "memory");
        bf16x16 pa = load_ab(lp + l16 * 32, half);

        // ---- O += P @ V  (V tile hd-major in LDS, keys contiguous) ----
#pragma unroll
        for (int t = 0; t < 4; ++t) {
            acc[t] = wmma_bf16(pa, load_ab(ldsV + (t * 16 + l16) * 32, half),
                               acc[t]);
        }
    }

    // ---- normalize and store bf16 [B,N,OUT] with head interleave ----
#pragma unroll
    for (int j = 0; j < 8; ++j) {
        float inv = 1.0f / lrow[j];
        size_t row = (size_t)b * NN + qbase + half * 8 + j;
#pragma unroll
        for (int t = 0; t < 4; ++t) {
            Oout[row * OUTD + h * HDD + t * 16 + l16] = (__bf16)(acc[t][j] * inv);
        }
    }
}

// ---------------------------------------------------------------------------
// Host-side orchestration
// ---------------------------------------------------------------------------
extern "C" void kernel_launch(void* const* d_in, const int* in_sizes, int n_in,
                              void* d_out, int out_size, void* d_ws, size_t ws_size,
                              hipStream_t stream) {
    const float* x   = (const float*)d_in[0];
    const int*   adj = (const int*)  d_in[1];
    const float* Wq  = (const float*)d_in[2];
    const float* bq  = (const float*)d_in[3];
    const float* Wk  = (const float*)d_in[4];
    const float* bk  = (const float*)d_in[5];
    const float* Wv  = (const float*)d_in[6];
    const float* bv  = (const float*)d_in[7];
    const float* Wo  = (const float*)d_in[8];
    const float* bo  = (const float*)d_in[9];

    // Workspace carve-up (all bf16)
    char* w = (char*)d_ws;
    __bf16* xb  = (__bf16*)w;  w += (size_t)MTOT * DD * 2;        // 8 MB
    __bf16* Wqt = (__bf16*)w;  w += (size_t)OUTD * DD * 2;        // 0.5 MB
    __bf16* Wkt = (__bf16*)w;  w += (size_t)OUTD * DD * 2;
    __bf16* Wvt = (__bf16*)w;  w += (size_t)OUTD * DD * 2;
    __bf16* Wot = (__bf16*)w;  w += (size_t)OUTD * OUTD * 2;
    __bf16* Qh  = (__bf16*)w;  w += (size_t)BB * HH * NN * HDD * 2;  // 8 MB
    __bf16* Kh  = (__bf16*)w;  w += (size_t)BB * HH * NN * HDD * 2;
    __bf16* Vth = (__bf16*)w;  w += (size_t)BB * HH * HDD * NN * 2;
    __bf16* AO  = (__bf16*)w;  w += (size_t)MTOT * OUTD * 2;

    // 1) precision conversion / weight transpose
    {
        int n = MTOT * DD;
        cvt_f32_bf16<<<(n + 255) / 256, 256, 0, stream>>>(x, xb, n);
        int nw = DD * OUTD;
        cvt_transpose_bf16<<<(nw + 255) / 256, 256, 0, stream>>>(Wq, Wqt, DD, OUTD);
        cvt_transpose_bf16<<<(nw + 255) / 256, 256, 0, stream>>>(Wk, Wkt, DD, OUTD);
        cvt_transpose_bf16<<<(nw + 255) / 256, 256, 0, stream>>>(Wv, Wvt, DD, OUTD);
        cvt_transpose_bf16<<<(nw + 255) / 256, 256, 0, stream>>>(Wo, Wot, OUTD, OUTD);
    }

    // 2) QKV projections (WMMA bf16)
    dim3 gg(MTOT / 64, OUTD / 32);
    gemm512_bf16<<<gg, 256, 0, stream>>>(xb, Wqt, bq, Qh, 0);
    gemm512_bf16<<<gg, 256, 0, stream>>>(xb, Wkt, bk, Kh, 0);
    gemm512_bf16<<<gg, 256, 0, stream>>>(xb, Wvt, bv, Vth, 1);

    // 3) fused masked flash attention (async-LDS double buffered K/V)
    dim3 ga(NN / 16 / 4, HH, BB);
    flash_attn_bf16<<<ga, 128, 0, stream>>>(Qh, Kh, Vth, adj, AO);

    // 4) output projection (fp32 out + bias)
    gemm512_bf16<<<gg, 256, 0, stream>>>(AO, Wot, bo, d_out, 2);
}
